// JITOptimizedMOE_67242007986682
// MI455X (gfx1250) — compile-verified, matrix operand order checked by reading, not run on hardware
//
#include <hip/hip_runtime.h>
#include <hip/hip_bf16.h>

// ---------------------------------------------------------------------------
// MoE (top-2 of 16 experts, F=512) + shared SwiGLU expert (F=1024)
// B=2, S=2048 -> T=4096 tokens, H=1024. fp32 in/out, bf16 WMMA compute.
// A-tiles staged with CDNA5 async global->LDS copies (ASYNCcnt).
// ---------------------------------------------------------------------------

typedef __attribute__((ext_vector_type(16))) __bf16 v16bf;
typedef __attribute__((ext_vector_type(8)))  float  v8f;

namespace moecfg {
constexpr int kT  = 4096;   // tokens
constexpr int kH  = 1024;   // hidden
constexpr int kE  = 16;     // experts
constexpr int kK  = 2;      // top-k
constexpr int kF  = 512;    // expert ffn
constexpr int kFS = 1024;   // shared ffn
constexpr int BM = 64, BN = 64, BK = 32;
}
using namespace moecfg;

// float -> bf16 via hardware convert (v_cvt_*bf16_f32), RNE
__device__ inline unsigned short f2bf(float f) {
  __bf16 h = (__bf16)f;
  return __builtin_bit_cast(unsigned short, h);
}

// CDNA5 async copy: 16 bytes global -> LDS per lane, tracked on ASYNCcnt.
__device__ inline void async_load_b128(const void* gptr, void* lds_ptr) {
  unsigned lds = (unsigned)(uintptr_t)lds_ptr;  // LDS aperture: addr[31:0] = LDS offset
  asm volatile("global_load_async_to_lds_b128 %0, %1, off"
               :: "v"(lds), "v"(gptr) : "memory");
}
__device__ inline void wait_async0() {
  asm volatile("s_wait_asynccnt 0x0" ::: "memory");
}

// Load a 16x32 bf16 fragment from LDS laid out row-major [rows][32].
// Lanes 0-15: row = lane,    K = 0..7 and 16..23
// Lanes 16-31: row = lane-16, K = 8..15 and 24..31   (CDNA5 16-bit A layout)
__device__ inline v16bf load_frag(const unsigned short* base, int row0, int lane) {
  const unsigned short* p = base + (row0 + (lane & 15)) * 32 + (lane >> 4) * 8;
  union { uint4 q[2]; v16bf v; } u;
  u.q[0] = *(const uint4*)(p);        // 8 bf16: K..K+7
  u.q[1] = *(const uint4*)(p + 16);   // 8 bf16: K+16..K+23
  return u.v;
}

__device__ inline v8f wmma_bf16(v16bf a, v16bf b, v8f c) {
  return __builtin_amdgcn_wmma_f32_16x16x32_bf16(false, a, false, b, (short)0, c,
                                                 false, false);
}

__device__ inline float silu(float g) { return g / (1.0f + __expf(-g)); }

// ---------------------------------------------------------------------------
// 0) zero per-expert counters
// ---------------------------------------------------------------------------
__global__ void moe_init(int* counts) {
  if (threadIdx.x < kE) counts[threadIdx.x] = 0;
}

// ---------------------------------------------------------------------------
// 1) x (fp32) -> xb (bf16), 8 elements/thread
// ---------------------------------------------------------------------------
__global__ __launch_bounds__(256) void moe_cast_x(const float* __restrict__ x,
                                                  unsigned short* __restrict__ xb) {
  int i = blockIdx.x * 256 + threadIdx.x;  // handles 8 floats
  const float4* s = (const float4*)x;
  float4 a = s[2 * i], b = s[2 * i + 1];
  union { unsigned short h[8]; uint4 q; } u;
  u.h[0] = f2bf(a.x); u.h[1] = f2bf(a.y); u.h[2] = f2bf(a.z); u.h[3] = f2bf(a.w);
  u.h[4] = f2bf(b.x); u.h[5] = f2bf(b.y); u.h[6] = f2bf(b.z); u.h[7] = f2bf(b.w);
  ((uint4*)xb)[i] = u.q;
}

// ---------------------------------------------------------------------------
// 2) router: logits -> softmax -> top2, one token per thread
// ---------------------------------------------------------------------------
__global__ __launch_bounds__(256) void moe_router(const float* __restrict__ x,
                                                  const float* __restrict__ rw,
                                                  int* counts, int* top_idx,
                                                  float* top_p) {
  int t = blockIdx.x * 256 + threadIdx.x;
  float acc[kE];
#pragma unroll
  for (int e = 0; e < kE; ++e) acc[e] = 0.0f;
  const float* xt = x + (size_t)t * kH;
  for (int h = 0; h < kH; ++h) {
    float xv = xt[h];
    const float4* r = (const float4*)(rw + (size_t)h * kE);
    float4 r0 = r[0], r1 = r[1], r2 = r[2], r3 = r[3];
    acc[0]  += xv * r0.x; acc[1]  += xv * r0.y; acc[2]  += xv * r0.z; acc[3]  += xv * r0.w;
    acc[4]  += xv * r1.x; acc[5]  += xv * r1.y; acc[6]  += xv * r1.z; acc[7]  += xv * r1.w;
    acc[8]  += xv * r2.x; acc[9]  += xv * r2.y; acc[10] += xv * r2.z; acc[11] += xv * r2.w;
    acc[12] += xv * r3.x; acc[13] += xv * r3.y; acc[14] += xv * r3.z; acc[15] += xv * r3.w;
  }
  float mx = acc[0];
#pragma unroll
  for (int e = 1; e < kE; ++e) mx = fmaxf(mx, acc[e]);
  float sum = 0.0f;
#pragma unroll
  for (int e = 0; e < kE; ++e) sum += __expf(acc[e] - mx);
  int i1 = 0; float l1 = acc[0];
#pragma unroll
  for (int e = 1; e < kE; ++e) if (acc[e] > l1) { l1 = acc[e]; i1 = e; }
  int i2 = (i1 == 0) ? 1 : 0; float l2 = acc[i2];
#pragma unroll
  for (int e = 0; e < kE; ++e)
    if (e != i1 && acc[e] > l2) { l2 = acc[e]; i2 = e; }
  float inv = 1.0f / sum;
  top_idx[t * 2 + 0] = i1; top_p[t * 2 + 0] = __expf(l1 - mx) * inv;
  top_idx[t * 2 + 1] = i2; top_p[t * 2 + 1] = __expf(l2 - mx) * inv;
  atomicAdd(&counts[i1], 1);
  atomicAdd(&counts[i2], 1);
}

// ---------------------------------------------------------------------------
// 3) prefix sum over E=16 counts
// ---------------------------------------------------------------------------
__global__ void moe_scan(const int* counts, int* offsets, int* cursor) {
  if (threadIdx.x == 0) {
    int off = 0;
    for (int e = 0; e < kE; ++e) { offsets[e] = off; cursor[e] = off; off += counts[e]; }
    offsets[kE] = off;
  }
}

// ---------------------------------------------------------------------------
// 4) compact per-expert token lists (one token per thread)
// ---------------------------------------------------------------------------
__global__ __launch_bounds__(256) void moe_fill(const int* top_idx, const float* top_p,
                                                int* cursor, int* tok_list,
                                                float* prob_list, int* slot_of) {
  int t = blockIdx.x * 256 + threadIdx.x;
#pragma unroll
  for (int k = 0; k < kK; ++k) {
    int e = top_idx[t * 2 + k];
    int slot = atomicAdd(&cursor[e], 1);
    tok_list[slot]  = t;
    prob_list[slot] = top_p[t * 2 + k];
    slot_of[t * 2 + k] = slot;
  }
}

// ---------------------------------------------------------------------------
// 5) shared expert gate/up:  inter_s = silu(x@Wg) * (x@Wu)   [T, FS] bf16
//    block tile 64x64, BK=32, 8 waves x 2 (gate+up) x 2 tiles
// ---------------------------------------------------------------------------
__global__ __launch_bounds__(256) void moe_shared_gateup(
    const unsigned short* __restrict__ xb, const float* __restrict__ gw,
    const float* __restrict__ uw, unsigned short* __restrict__ inter_s) {
  __shared__ unsigned short As[BM * BK], Bg[BN * BK], Bu[BN * BK];
  int m0 = blockIdx.x * BM, n0 = blockIdx.y * BN;
  int tid = threadIdx.x, lane = tid & 31, wave = tid >> 5;
  v8f ag[2] = {}, au[2] = {};
  for (int k0 = 0; k0 < kH; k0 += BK) {
    __syncthreads();
    {  // A: 64 rows x 32 bf16, async 16B per lane (no conversion needed)
      int row = tid >> 2, q = tid & 3;
      async_load_b128(xb + (size_t)(m0 + row) * kH + k0 + q * 8, As + tid * 8);
    }
    {  // B: transpose [k][n] fp32 -> LDS [n][32k] bf16 (hardware cvt)
      int kk = tid >> 3, nb = (tid & 7) * 8;
#pragma unroll
      for (int j = 0; j < 8; j += 2) {
        float2 g2 = *(const float2*)(gw + (size_t)(k0 + kk) * kFS + n0 + nb + j);
        Bg[(nb + j) * 32 + kk] = f2bf(g2.x);
        Bg[(nb + j + 1) * 32 + kk] = f2bf(g2.y);
        float2 u2 = *(const float2*)(uw + (size_t)(k0 + kk) * kFS + n0 + nb + j);
        Bu[(nb + j) * 32 + kk] = f2bf(u2.x);
        Bu[(nb + j + 1) * 32 + kk] = f2bf(u2.y);
      }
    }
    if (tid == 0 && k0 + BK < kH) {
      __builtin_prefetch(gw + (size_t)(k0 + BK) * kFS + n0, 0, 1);
      __builtin_prefetch(uw + (size_t)(k0 + BK) * kFS + n0, 0, 1);
    }
    wait_async0();
    __syncthreads();
#pragma unroll
    for (int j = 0; j < 2; ++j) {
      int idx = wave * 2 + j, mi = idx >> 2, ni = idx & 3;
      v16bf a  = load_frag(As, mi * 16, lane);
      v16bf bg = load_frag(Bg, ni * 16, lane);
      ag[j] = wmma_bf16(a, bg, ag[j]);
      v16bf bu = load_frag(Bu, ni * 16, lane);
      au[j] = wmma_bf16(a, bu, au[j]);
    }
  }
#pragma unroll
  for (int j = 0; j < 2; ++j) {
    int idx = wave * 2 + j, mi = idx >> 2, ni = idx & 3;
    int n = n0 + ni * 16 + (lane & 15);
    int mb = m0 + mi * 16 + (lane >> 4) * 8;
#pragma unroll
    for (int i = 0; i < 8; ++i) {
      float h = silu(ag[j][i]) * au[j][i];
      inter_s[(size_t)(mb + i) * kFS + n] = f2bf(h);
    }
  }
}

// ---------------------------------------------------------------------------
// 6) shared expert down: out = inter_s @ Wd   (plain store -> initializes out)
// ---------------------------------------------------------------------------
__global__ __launch_bounds__(256) void moe_shared_down(
    const unsigned short* __restrict__ inter_s, const float* __restrict__ dw,
    float* __restrict__ out) {
  __shared__ unsigned short As[BM * BK], Bd[BN * BK];
  int m0 = blockIdx.x * BM, n0 = blockIdx.y * BN;
  int tid = threadIdx.x, lane = tid & 31, wave = tid >> 5;
  v8f ac[2] = {};
  for (int k0 = 0; k0 < kFS; k0 += BK) {
    __syncthreads();
    {
      int row = tid >> 2, q = tid & 3;
      async_load_b128(inter_s + (size_t)(m0 + row) * kFS + k0 + q * 8, As + tid * 8);
    }
    {
      int kk = tid >> 3, nb = (tid & 7) * 8;
#pragma unroll
      for (int j = 0; j < 8; j += 2) {
        float2 d2 = *(const float2*)(dw + (size_t)(k0 + kk) * kH + n0 + nb + j);
        Bd[(nb + j) * 32 + kk] = f2bf(d2.x);
        Bd[(nb + j + 1) * 32 + kk] = f2bf(d2.y);
      }
    }
    if (tid == 0 && k0 + BK < kFS)
      __builtin_prefetch(dw + (size_t)(k0 + BK) * kH + n0, 0, 1);
    wait_async0();
    __syncthreads();
#pragma unroll
    for (int j = 0; j < 2; ++j) {
      int idx = wave * 2 + j, mi = idx >> 2, ni = idx & 3;
      v16bf a = load_frag(As, mi * 16, lane);
      v16bf b = load_frag(Bd, ni * 16, lane);
      ac[j] = wmma_bf16(a, b, ac[j]);
    }
  }
#pragma unroll
  for (int j = 0; j < 2; ++j) {
    int idx = wave * 2 + j, mi = idx >> 2, ni = idx & 3;
    int n = n0 + ni * 16 + (lane & 15);
    int mb = m0 + mi * 16 + (lane >> 4) * 8;
#pragma unroll
    for (int i = 0; i < 8; ++i)
      out[(size_t)(mb + i) * kH + n] = ac[j][i];
  }
}

// ---------------------------------------------------------------------------
// 7) routed gate/up over gathered tokens: inter_r[slot] = silu(x@g)*(x@u)
//    grid: (m-tiles, F/64, expert)
// ---------------------------------------------------------------------------
__global__ __launch_bounds__(256) void moe_routed_gateup(
    const unsigned short* __restrict__ xb, const float* __restrict__ gw,
    const float* __restrict__ uw, const int* counts, const int* offsets,
    const int* tok_list, unsigned short* __restrict__ inter_r) {
  int e = blockIdx.z;
  int cnt = counts[e];
  int m0 = blockIdx.x * BM;
  if (m0 >= cnt) return;
  int seg = offsets[e];
  int n0 = blockIdx.y * BN;
  const float* gwe = gw + (size_t)e * kH * kF;
  const float* uwe = uw + (size_t)e * kH * kF;
  __shared__ unsigned short As[BM * BK], Bg[BN * BK], Bu[BN * BK];
  __shared__ int toks[BM];
  int tid = threadIdx.x, lane = tid & 31, wave = tid >> 5;
  if (tid < BM) {
    int r = m0 + tid;
    if (r >= cnt) r = cnt - 1;
    toks[tid] = tok_list[seg + r];
  }
  v8f ag[2] = {}, au[2] = {};
  for (int k0 = 0; k0 < kH; k0 += BK) {
    __syncthreads();
    {  // gathered A rows, async 16B per lane
      int row = tid >> 2, q = tid & 3;
      async_load_b128(xb + (size_t)toks[row] * kH + k0 + q * 8, As + tid * 8);
    }
    {
      int kk = tid >> 3, nb = (tid & 7) * 8;
#pragma unroll
      for (int j = 0; j < 8; j += 2) {
        float2 g2 = *(const float2*)(gwe + (size_t)(k0 + kk) * kF + n0 + nb + j);
        Bg[(nb + j) * 32 + kk] = f2bf(g2.x);
        Bg[(nb + j + 1) * 32 + kk] = f2bf(g2.y);
        float2 u2 = *(const float2*)(uwe + (size_t)(k0 + kk) * kF + n0 + nb + j);
        Bu[(nb + j) * 32 + kk] = f2bf(u2.x);
        Bu[(nb + j + 1) * 32 + kk] = f2bf(u2.y);
      }
    }
    if (tid == 0 && k0 + BK < kH) {
      __builtin_prefetch(gwe + (size_t)(k0 + BK) * kF + n0, 0, 1);
      __builtin_prefetch(uwe + (size_t)(k0 + BK) * kF + n0, 0, 1);
    }
    wait_async0();
    __syncthreads();
#pragma unroll
    for (int j = 0; j < 2; ++j) {
      int idx = wave * 2 + j, mi = idx >> 2, ni = idx & 3;
      v16bf a  = load_frag(As, mi * 16, lane);
      v16bf bg = load_frag(Bg, ni * 16, lane);
      ag[j] = wmma_bf16(a, bg, ag[j]);
      v16bf bu = load_frag(Bu, ni * 16, lane);
      au[j] = wmma_bf16(a, bu, au[j]);
    }
  }
#pragma unroll
  for (int j = 0; j < 2; ++j) {
    int idx = wave * 2 + j, mi = idx >> 2, ni = idx & 3;
    int n = n0 + ni * 16 + (lane & 15);
    int mb = m0 + mi * 16 + (lane >> 4) * 8;
#pragma unroll
    for (int i = 0; i < 8; ++i) {
      int m = mb + i;
      if (m < cnt) {
        float h = silu(ag[j][i]) * au[j][i];
        inter_r[(size_t)(seg + m) * kF + n] = f2bf(h);
      }
    }
  }
}

// ---------------------------------------------------------------------------
// 8) routed down: rout[slot] = prob[slot] * (inter_r[slot] @ Wd[e])
// ---------------------------------------------------------------------------
__global__ __launch_bounds__(256) void moe_routed_down(
    const unsigned short* __restrict__ inter_r, const float* __restrict__ dw,
    const int* counts, const int* offsets, const float* prob_list,
    float* __restrict__ rout) {
  int e = blockIdx.z;
  int cnt = counts[e];
  int m0 = blockIdx.x * BM;
  if (m0 >= cnt) return;
  int seg = offsets[e];
  int n0 = blockIdx.y * BN;
  const float* dwe = dw + (size_t)e * kF * kH;
  __shared__ unsigned short As[BM * BK], Bd[BN * BK];
  int tid = threadIdx.x, lane = tid & 31, wave = tid >> 5;
  v8f ac[2] = {};
  for (int k0 = 0; k0 < kF; k0 += BK) {
    __syncthreads();
    {
      int row = tid >> 2, q = tid & 3;
      int r = m0 + row;
      if (r >= cnt) r = cnt - 1;
      async_load_b128(inter_r + (size_t)(seg + r) * kF + k0 + q * 8, As + tid * 8);
    }
    {
      int kk = tid >> 3, nb = (tid & 7) * 8;
#pragma unroll
      for (int j = 0; j < 8; j += 2) {
        float2 d2 = *(const float2*)(dwe + (size_t)(k0 + kk) * kH + n0 + nb + j);
        Bd[(nb + j) * 32 + kk] = f2bf(d2.x);
        Bd[(nb + j + 1) * 32 + kk] = f2bf(d2.y);
      }
    }
    if (tid == 0 && k0 + BK < kF)
      __builtin_prefetch(dwe + (size_t)(k0 + BK) * kH + n0, 0, 1);
    wait_async0();
    __syncthreads();
#pragma unroll
    for (int j = 0; j < 2; ++j) {
      int idx = wave * 2 + j, mi = idx >> 2, ni = idx & 3;
      v16bf a = load_frag(As, mi * 16, lane);
      v16bf b = load_frag(Bd, ni * 16, lane);
      ac[j] = wmma_bf16(a, b, ac[j]);
    }
  }
#pragma unroll
  for (int j = 0; j < 2; ++j) {
    int idx = wave * 2 + j, mi = idx >> 2, ni = idx & 3;
    int n = n0 + ni * 16 + (lane & 15);
    int mb = m0 + mi * 16 + (lane >> 4) * 8;
#pragma unroll
    for (int i = 0; i < 8; ++i) {
      int m = mb + i;
      if (m < cnt) {
        float p = prob_list[seg + m];
        rout[(size_t)(seg + m) * kH + n] = ac[j][i] * p;
      }
    }
  }
}

// ---------------------------------------------------------------------------
// 9) combine: out[t] += rout[slot0(t)] + rout[slot1(t)]  (race-free, 4 f32/thread)
// ---------------------------------------------------------------------------
__global__ __launch_bounds__(256) void moe_combine(float* __restrict__ out,
                                                   const float* __restrict__ rout,
                                                   const int* __restrict__ slot_of) {
  int gi = blockIdx.x * 256 + threadIdx.x;  // over T*H/4
  int t = gi / (kH / 4);
  int c = gi % (kH / 4);
  int s0 = slot_of[t * 2 + 0], s1 = slot_of[t * 2 + 1];
  const float4* r0 = (const float4*)(rout + (size_t)s0 * kH);
  const float4* r1 = (const float4*)(rout + (size_t)s1 * kH);
  float4* o = (float4*)(out + (size_t)t * kH);
  float4 a = o[c], b = r0[c], d = r1[c];
  a.x += b.x + d.x; a.y += b.y + d.y; a.z += b.z + d.z; a.w += b.w + d.w;
  o[c] = a;
}

// ---------------------------------------------------------------------------
// launch
// ---------------------------------------------------------------------------
extern "C" void kernel_launch(void* const* d_in, const int* in_sizes, int n_in,
                              void* d_out, int out_size, void* d_ws, size_t ws_size,
                              hipStream_t stream) {
  (void)in_sizes; (void)n_in; (void)out_size; (void)ws_size;
  const float* x   = (const float*)d_in[0];
  const float* rw  = (const float*)d_in[1];
  const float* gw  = (const float*)d_in[2];
  const float* uw  = (const float*)d_in[3];
  const float* dw  = (const float*)d_in[4];
  const float* sgw = (const float*)d_in[5];
  const float* suw = (const float*)d_in[6];
  const float* sdw = (const float*)d_in[7];
  float* out = (float*)d_out;

  char* p = (char*)d_ws;
  auto take = [&](size_t bytes) {
    char* cur = p;
    p += (bytes + 255) & ~size_t(255);
    return cur;
  };
  int*   counts    = (int*)take(kE * sizeof(int));
  int*   cursor    = (int*)take(kE * sizeof(int));
  int*   offsets   = (int*)take((kE + 1) * sizeof(int));
  int*   top_idx   = (int*)take((size_t)kT * kK * sizeof(int));
  float* top_p     = (float*)take((size_t)kT * kK * sizeof(float));
  int*   tok_list  = (int*)take((size_t)kT * kK * sizeof(int));
  float* prob_list = (float*)take((size_t)kT * kK * sizeof(float));
  int*   slot_of   = (int*)take((size_t)kT * kK * sizeof(int));
  unsigned short* xb      = (unsigned short*)take((size_t)kT * kH * 2);
  unsigned short* inter_s = (unsigned short*)take((size_t)kT * kFS * 2);
  unsigned short* inter_r = (unsigned short*)take((size_t)kT * kK * kF * 2);
  float*          rout    = (float*)take((size_t)kT * kK * kH * sizeof(float));

  moe_init<<<1, 32, 0, stream>>>(counts);
  moe_cast_x<<<(kT * kH / 8) / 256, 256, 0, stream>>>(x, xb);
  moe_router<<<kT / 256, 256, 0, stream>>>(x, rw, counts, top_idx, top_p);
  moe_scan<<<1, 32, 0, stream>>>(counts, offsets, cursor);
  moe_fill<<<kT / 256, 256, 0, stream>>>(top_idx, top_p, cursor, tok_list,
                                         prob_list, slot_of);
  moe_shared_gateup<<<dim3(kT / BM, kFS / BN), 256, 0, stream>>>(xb, sgw, suw,
                                                                 inter_s);
  moe_shared_down<<<dim3(kT / BM, kH / BN), 256, 0, stream>>>(inter_s, sdw, out);
  moe_routed_gateup<<<dim3(kT / BM, kF / BN, kE), 256, 0, stream>>>(
      xb, gw, uw, counts, offsets, tok_list, inter_r);
  moe_routed_down<<<dim3(kT / BM, kH / BN, kE), 256, 0, stream>>>(
      inter_r, dw, counts, offsets, prob_list, rout);
  moe_combine<<<(kT * kH / 4) / 256, 256, 0, stream>>>(out, rout, slot_of);
}
